// HGT_39883066310773
// MI455X (gfx1250) — matrix operand order repaired and checked by reading, not compiled, not run on hardware
//
#include <hip/hip_runtime.h>
#include <math.h>

#define N_NODES 50000
#define N_EDGES 250000
#define N_REL   6
#define IN_FEAT 1280
#define HID     512
#define HEADS   8
#define DKH     64
#define N_LAYERS 2

typedef _Float16 half_t;
typedef half_t v16h __attribute__((ext_vector_type(16)));
typedef float  v8f  __attribute__((ext_vector_type(8)));

union Frag16 { v16h v; uint4 u[2]; };

__device__ __forceinline__ void atomicMaxF32(float* addr, float val) {
    // monotonic int/uint trick; buffer initialized to -inf (0xFF800000)
    if (val >= 0.0f) atomicMax((int*)addr, __float_as_int(val));
    else             atomicMin((unsigned int*)addr, __float_as_uint(val));
}

// ---------------------------------------------------------------------------
// WMMA GEMM, register-blocked 16x64 per wave:
//   C[M,Nc] = A[M,K] (f16, row-major, stride lda) @ Wt ([Nc][K] f16, stride ldk)
//           + bias[Nc] (optional)
// One wave -> 4 adjacent 16x16 tiles (64 columns): A fragment loaded once per
// K-step and reused across 4 WMMAs. block = (32,4): 4 waves = 256 columns.
// M = N_NODES (mult of 16). K mult of 32. Nc mult of 64.
// Guard is wave-uniform, so EXEC is all-1s for every wave reaching WMMA.
// ---------------------------------------------------------------------------
__global__ __launch_bounds__(128)
void hgt_wmma_gemm(const half_t* __restrict__ A, int lda,
                   const half_t* __restrict__ Wt, int ldk,
                   const float* __restrict__ bias,
                   float* __restrict__ C, int ldc, int Nc, int Ktot)
{
    const int lane   = threadIdx.x & 31;
    const int warp   = threadIdx.y;
    const int tileM  = blockIdx.x;
    const int tileN4 = blockIdx.y * 4 + warp;   // 64-column strip index
    if (tileN4 * 64 >= Nc) return;              // wave-uniform exit

    const int hi  = lane >> 4;   // half-wave select
    const int l15 = lane & 15;

    // A fragment (16x32 f16): lanes 0-15 row=l15, elems 0..7=K0..7, 8..15=K16..23
    //                         lanes 16-31 same rows, K offset +8 / +24
    const half_t* ap = A + (long)(tileM * 16 + l15) * lda + hi * 8;

    const int n0 = tileN4 * 64 + l15;           // this lane's column in strip j=0
    v8f acc[4];
#pragma unroll
    for (int j = 0; j < 4; ++j) {
        const float bv = bias ? bias[n0 + j * 16] : 0.0f;
#pragma unroll
        for (int e = 0; e < 8; ++e) acc[j][e] = bv;
    }

    // B fragment (32x16 f16) from Wt[n][k]: lane l15 = column, K order same as A
    const half_t* bp = Wt + (long)n0 * ldk + hi * 8;

    for (int k0 = 0; k0 < Ktot; k0 += 32) {
        Frag16 af;
        af.u[0] = *(const uint4*)(ap + k0);
        af.u[1] = *(const uint4*)(ap + k0 + 16);
#pragma unroll
        for (int j = 0; j < 4; ++j) {
            const half_t* bj = bp + (long)j * 16 * ldk + k0;
            Frag16 bf;
            bf.u[0] = *(const uint4*)(bj);
            bf.u[1] = *(const uint4*)(bj + 16);
            acc[j] = __builtin_amdgcn_wmma_f32_16x16x32_f16(
                         false, af.v, false, bf.v, (short)0, acc[j], false, false);
        }
    }

    // C/D layout: VGPR v holds (row = v + 8*hi, col = lane&15)
    float* cbase = C + (long)(tileM * 16 + hi * 8) * ldc + n0;
#pragma unroll
    for (int j = 0; j < 4; ++j)
#pragma unroll
        for (int v = 0; v < 8; ++v)
            cbase[(long)v * ldc + j * 16] = acc[j][v];
}

// ---------------------------------------------------------------------------
// small utility kernels
// ---------------------------------------------------------------------------
__global__ void hgt_f32_to_f16(const float* __restrict__ x, half_t* __restrict__ y, long n) {
    long i = (long)blockIdx.x * blockDim.x + threadIdx.x;
    if (i < n) y[i] = (half_t)x[i];
}

__global__ void hgt_gelu_to_f16(const float* __restrict__ x, half_t* __restrict__ y, long n) {
    long i = (long)blockIdx.x * blockDim.x + threadIdx.x;
    if (i < n) {
        float v = x[i];
        y[i] = (half_t)(0.5f * v * (1.0f + erff(v * 0.70710678118654752f)));
    }
}

__global__ void hgt_fill_f32(float* __restrict__ p, float v, long n) {
    long i = (long)blockIdx.x * blockDim.x + threadIdx.x;
    if (i < n) p[i] = v;
}

// W[K][Nc] f32 -> Wt[Nc][K] f16
__global__ void hgt_transpose_w(const float* __restrict__ W, half_t* __restrict__ Wt,
                                int K, int Nc) {
    long i = (long)blockIdx.x * blockDim.x + threadIdx.x;
    if (i >= (long)K * Nc) return;
    int nn = (int)(i / K), kk = (int)(i % K);
    Wt[i] = (half_t)W[(long)kk * Nc + nn];
}

// rel matrix [H][d][f] f32 -> Wt[H][f][d] f16  (per-head transpose)
__global__ void hgt_transpose_heads(const float* __restrict__ Wsrc, half_t* __restrict__ Wt) {
    int i = blockIdx.x * blockDim.x + threadIdx.x;
    if (i >= HEADS * DKH * DKH) return;
    int hh = i >> 12, rem = i & 4095, f = rem >> 6, d = rem & 63;
    Wt[i] = (half_t)Wsrc[hh * 4096 + d * 64 + f];
}

// ---------------------------------------------------------------------------
// edge-level attention kernels (one thread per edge*head)
// ---------------------------------------------------------------------------
__global__ void hgt_edge_score(const float* __restrict__ q, const float* __restrict__ k_r,
                               const int* __restrict__ src, const int* __restrict__ dst,
                               const float* __restrict__ pri,
                               float* __restrict__ score, float* __restrict__ mbuf)
{
    long i = (long)blockIdx.x * blockDim.x + threadIdx.x;
    if (i >= (long)N_EDGES * HEADS) return;
    int e = (int)(i >> 3), hh = (int)(i & 7);
    int s = src[e], d = dst[e];
    const float* qp = q   + (long)d * HID + hh * DKH;
    const float* kp = k_r + (long)s * HID + hh * DKH;
    float acc = 0.0f;
#pragma unroll 8
    for (int j = 0; j < DKH; ++j) acc += qp[j] * kp[j];
    acc *= pri[hh] * 0.125f;              // / sqrt(64)
    score[i] = acc;
    atomicMaxF32(mbuf + (long)d * HEADS + hh, acc);
}

__global__ void hgt_edge_exp(const int* __restrict__ dst, const float* __restrict__ mbuf,
                             float* __restrict__ score, float* __restrict__ ssum)
{
    long i = (long)blockIdx.x * blockDim.x + threadIdx.x;
    if (i >= (long)N_EDGES * HEADS) return;
    int e = (int)(i >> 3), hh = (int)(i & 7);
    int d = dst[e];
    float w = expf(score[i] - mbuf[(long)d * HEADS + hh]);
    score[i] = w;
    atomicAdd(ssum + (long)d * HEADS + hh, w);
}

__global__ void hgt_edge_agg(const int* __restrict__ src, const int* __restrict__ dst,
                             const float* __restrict__ score, const float* __restrict__ ssum,
                             const float* __restrict__ v_r, float* __restrict__ agg)
{
    long i = (long)blockIdx.x * blockDim.x + threadIdx.x;
    if (i >= (long)N_EDGES * HEADS) return;
    int e = (int)(i >> 3), hh = (int)(i & 7);
    int s = src[e], d = dst[e];
    float att = score[i] / (ssum[(long)d * HEADS + hh] + 1e-16f);
    const float* vp = v_r + (long)s * HID + hh * DKH;
    float*       ap = agg + (long)d * HID + hh * DKH;
#pragma unroll 4
    for (int j = 0; j < DKH; ++j) atomicAdd(ap + j, att * vp[j]);
}

// ---------------------------------------------------------------------------
// skip + LayerNorm: h = LN(trans*alpha + h*(1-alpha)); also emits f16 copy
// one block (256 threads) per node, 2 channels per thread
// ---------------------------------------------------------------------------
__global__ __launch_bounds__(256)
void hgt_skip_ln(const float* __restrict__ trans, float* __restrict__ h,
                 half_t* __restrict__ h16, const float* __restrict__ skip_p,
                 const float* __restrict__ g, const float* __restrict__ b)
{
    __shared__ float red[256];
    const int node = blockIdx.x;
    const int t = threadIdx.x;
    const float alpha = 1.0f / (1.0f + expf(-skip_p[0]));
    const float* tp = trans + (long)node * HID;
    float*       hp = h     + (long)node * HID;

    float o0 = tp[t]       * alpha + hp[t]       * (1.0f - alpha);
    float o1 = tp[t + 256] * alpha + hp[t + 256] * (1.0f - alpha);

    red[t] = o0 + o1;
    __syncthreads();
    for (int s = 128; s > 0; s >>= 1) { if (t < s) red[t] += red[t + s]; __syncthreads(); }
    float mu = red[0] * (1.0f / HID);
    __syncthreads();
    float d0 = o0 - mu, d1 = o1 - mu;
    red[t] = d0 * d0 + d1 * d1;
    __syncthreads();
    for (int s = 128; s > 0; s >>= 1) { if (t < s) red[t] += red[t + s]; __syncthreads(); }
    float rstd = rsqrtf(red[0] * (1.0f / HID) + 1e-5f);

    float r0 = d0 * rstd * g[t]       + b[t];
    float r1 = d1 * rstd * g[t + 256] + b[t + 256];
    hp[t] = r0;  hp[t + 256] = r1;
    h16[(long)node * HID + t]       = (half_t)r0;
    h16[(long)node * HID + t + 256] = (half_t)r1;
}

// ---------------------------------------------------------------------------
// readout: column means over nodes, then dot with Wc + bc -> scalar
// ---------------------------------------------------------------------------
__global__ __launch_bounds__(256)
void hgt_col_mean(const float* __restrict__ h, float* __restrict__ hg) {
    __shared__ float red[256];
    int c = blockIdx.x, t = threadIdx.x;
    float acc = 0.0f;
    for (int n = t; n < N_NODES; n += 256) acc += h[(long)n * HID + c];
    red[t] = acc; __syncthreads();
    for (int s = 128; s > 0; s >>= 1) { if (t < s) red[t] += red[t + s]; __syncthreads(); }
    if (t == 0) hg[c] = red[0] * (1.0f / N_NODES);
}

__global__ __launch_bounds__(256)
void hgt_final_dot(const float* __restrict__ hg, const float* __restrict__ Wc,
                   const float* __restrict__ bc, float* __restrict__ out) {
    __shared__ float red[256];
    int t = threadIdx.x;
    float acc = hg[t] * Wc[t] + hg[t + 256] * Wc[t + 256];
    red[t] = acc; __syncthreads();
    for (int s = 128; s > 0; s >>= 1) { if (t < s) red[t] += red[t + s]; __syncthreads(); }
    if (t == 0) out[0] = red[0] + bc[0];
}

// ---------------------------------------------------------------------------
// host orchestration
// ---------------------------------------------------------------------------
extern "C" void kernel_launch(void* const* d_in, const int* in_sizes, int n_in,
                              void* d_out, int out_size, void* d_ws, size_t ws_size,
                              hipStream_t stream)
{
    (void)in_sizes; (void)n_in; (void)out_size; (void)ws_size;

    const float* node_emb = (const float*)d_in[0];
    const int*   edge_src = (const int*)d_in[1];
    const int*   edge_dst = (const int*)d_in[2];
    const float* Wi = (const float*)d_in[3];
    const float* bi = (const float*)d_in[4];
    const float* Wc = (const float*)d_in[5];
    const float* bc = (const float*)d_in[6];
    // layers: base = 7 + l*14 : Wk bk Wq bq Wv bv Wa ba rel_pri rel_att rel_msg skip ln_g ln_b

    // ---- workspace carve (bump allocator) ----
    char* wsp = (char*)d_ws;
    auto carve = [&](size_t bytes) -> char* {
        char* r = wsp; wsp += (bytes + 255) & ~(size_t)255; return r;
    };
    const long NH  = (long)N_NODES * HID;
    float* h     = (float*)carve(sizeof(float) * NH);
    float* tmp   = (float*)carve(sizeof(float) * NH);
    float* qbuf  = (float*)carve(sizeof(float) * NH);
    float* k_r   = (float*)carve(sizeof(float) * NH);
    float* v_r   = (float*)carve(sizeof(float) * NH);
    float* agg   = (float*)carve(sizeof(float) * NH);
    float* mbuf  = (float*)carve(sizeof(float) * (long)N_NODES * HEADS);
    float* ssum  = (float*)carve(sizeof(float) * (long)N_NODES * HEADS);
    float* score = (float*)carve(sizeof(float) * (long)N_EDGES * HEADS);
    float* hg    = (float*)carve(sizeof(float) * HID);
    half_t* emb16 = (half_t*)carve(sizeof(half_t) * (long)N_NODES * IN_FEAT);
    half_t* h16   = (half_t*)carve(sizeof(half_t) * NH);
    half_t* x16   = (half_t*)carve(sizeof(half_t) * NH);
    half_t* WiT   = (half_t*)carve(sizeof(half_t) * (long)IN_FEAT * HID);
    half_t* WT[N_LAYERS][4];
    for (int l = 0; l < N_LAYERS; ++l)
        for (int j = 0; j < 4; ++j)
            WT[l][j] = (half_t*)carve(sizeof(half_t) * (long)HID * HID);
    half_t* attT = (half_t*)carve(sizeof(half_t) * HEADS * DKH * DKH);
    half_t* msgT = (half_t*)carve(sizeof(half_t) * HEADS * DKH * DKH);

    auto gemm = [&](const half_t* A, int lda, const half_t* Wt, int ldk,
                    const float* bias, float* C, int ldc, int Nc, int Ktot) {
        dim3 blk(32, 4), grd(N_NODES / 16, (unsigned)((Nc + 255) / 256));
        hgt_wmma_gemm<<<grd, blk, 0, stream>>>(A, lda, Wt, ldk, bias, C, ldc, Nc, Ktot);
    };

    // ---- one-time prep: input + weight f16 conversion/transpose ----
    {
        long ne = (long)N_NODES * IN_FEAT;
        hgt_f32_to_f16<<<(unsigned)((ne + 255) / 256), 256, 0, stream>>>(node_emb, emb16, ne);
        long wi = (long)IN_FEAT * HID;
        hgt_transpose_w<<<(unsigned)((wi + 255) / 256), 256, 0, stream>>>(Wi, WiT, IN_FEAT, HID);
        for (int l = 0; l < N_LAYERS; ++l)
            for (int j = 0; j < 4; ++j) {  // Wk, Wq, Wv, Wa at base+0,2,4,6
                const float* W = (const float*)d_in[7 + l * 14 + 2 * j];
                long n = (long)HID * HID;
                hgt_transpose_w<<<(unsigned)((n + 255) / 256), 256, 0, stream>>>(W, WT[l][j], HID, HID);
            }
    }

    // ---- input projection: h = emb @ Wi + bi ----
    gemm(emb16, IN_FEAT, WiT, IN_FEAT, bi, h, HID, HID, IN_FEAT);
    hgt_f32_to_f16<<<(unsigned)((NH + 255) / 256), 256, 0, stream>>>(h, h16, NH);

    const long EH = (long)N_EDGES * HEADS;
    const unsigned ehBlocks = (unsigned)((EH + 255) / 256);
    const unsigned nhBlocks = (unsigned)((NH + 255) / 256);
    const long NHh = (long)N_NODES * HEADS;

    for (int l = 0; l < N_LAYERS; ++l) {
        const int base = 7 + l * 14;
        const float* bk      = (const float*)d_in[base + 1];
        const float* bq      = (const float*)d_in[base + 3];
        const float* bv      = (const float*)d_in[base + 5];
        const float* ba      = (const float*)d_in[base + 7];
        const float* rel_pri = (const float*)d_in[base + 8];
        const float* rel_att = (const float*)d_in[base + 9];
        const float* rel_msg = (const float*)d_in[base + 10];
        const float* skip    = (const float*)d_in[base + 11];
        const float* ln_g    = (const float*)d_in[base + 12];
        const float* ln_b    = (const float*)d_in[base + 13];

        for (int r = 0; r < N_REL; ++r) {
            const int* src = edge_src + (long)r * N_EDGES;
            const int* dst = edge_dst + (long)r * N_EDGES;

            hgt_transpose_heads<<<HEADS * DKH * DKH / 256, 256, 0, stream>>>(
                rel_att + (long)r * HEADS * DKH * DKH, attT);
            hgt_transpose_heads<<<HEADS * DKH * DKH / 256, 256, 0, stream>>>(
                rel_msg + (long)r * HEADS * DKH * DKH, msgT);

            // K = h @ Wk + bk ; then per-head k_r = K_h @ rel_att[r][h]
            gemm(h16, HID, WT[l][0], HID, bk, tmp, HID, HID, HID);
            hgt_f32_to_f16<<<nhBlocks, 256, 0, stream>>>(tmp, x16, NH);
            for (int hh = 0; hh < HEADS; ++hh)
                gemm(x16 + hh * DKH, HID, attT + hh * DKH * DKH, DKH,
                     nullptr, k_r + hh * DKH, HID, DKH, DKH);

            // V = h @ Wv + bv ; then per-head v_r = V_h @ rel_msg[r][h]
            gemm(h16, HID, WT[l][2], HID, bv, tmp, HID, HID, HID);
            hgt_f32_to_f16<<<nhBlocks, 256, 0, stream>>>(tmp, x16, NH);
            for (int hh = 0; hh < HEADS; ++hh)
                gemm(x16 + hh * DKH, HID, msgT + hh * DKH * DKH, DKH,
                     nullptr, v_r + hh * DKH, HID, DKH, DKH);

            // Q = h @ Wq + bq
            gemm(h16, HID, WT[l][1], HID, bq, qbuf, HID, HID, HID);

            // segment softmax + aggregation
            hgt_fill_f32<<<(unsigned)((NHh + 255) / 256), 256, 0, stream>>>(mbuf, -INFINITY, NHh);
            hgt_fill_f32<<<(unsigned)((NHh + 255) / 256), 256, 0, stream>>>(ssum, 0.0f, NHh);
            hgt_fill_f32<<<nhBlocks, 256, 0, stream>>>(agg, 0.0f, NH);

            hgt_edge_score<<<ehBlocks, 256, 0, stream>>>(qbuf, k_r, src, dst,
                                                         rel_pri + r * HEADS, score, mbuf);
            hgt_edge_exp<<<ehBlocks, 256, 0, stream>>>(dst, mbuf, score, ssum);
            hgt_edge_agg<<<ehBlocks, 256, 0, stream>>>(src, dst, score, ssum, v_r, agg);

            // trans = gelu(agg) @ Wa + ba ; then skip + LayerNorm -> h, h16
            hgt_gelu_to_f16<<<nhBlocks, 256, 0, stream>>>(agg, x16, NH);
            gemm(x16, HID, WT[l][3], HID, ba, tmp, HID, HID, HID);
            hgt_skip_ln<<<N_NODES, 256, 0, stream>>>(tmp, h, h16, skip, ln_g, ln_b);
        }
    }

    // ---- readout ----
    hgt_col_mean<<<HID, 256, 0, stream>>>(h, hg);
    hgt_final_dot<<<1, 256, 0, stream>>>(hg, Wc, bc, (float*)d_out);
}
